// EmbeddingVicent_77111842832399
// MI455X (gfx1250) — matrix operand order repaired and checked by reading, not compiled, not run on hardware
//
#include <hip/hip_runtime.h>
#include <hip/hip_bf16.h>

typedef __attribute__((ext_vector_type(16))) _Float16 v16h;
typedef __attribute__((ext_vector_type(8)))  float    v8f;

#define LN_EPS 1e-5f

// Fused: embedding-gather + scalar-affine feature expansion + [N,28]x[28,64]
// GEMM (f16 WMMA, f32 accum, bias folded into WMMA C operand) + PReLU +
// LayerNorm(64). One wave32 handles a 16-row tile; 4 x
// v_wmma_f32_16x16x32_f16 per tile (N split into 4 tiles of 16 cols,
// K=28 zero-padded to 32).
__global__ __launch_bounds__(256) void fused_embed_dense_prelu_ln(
    const int*   __restrict__ notes,       // [N]
    const float* __restrict__ onsets,      // [N]
    const float* __restrict__ durs,        // [N]
    const float* __restrict__ note_table,  // [91,16]
    const float* __restrict__ W_on,        // [8]
    const float* __restrict__ b_on,        // [8]
    const float* __restrict__ W_dur,       // [4]
    const float* __restrict__ b_dur,       // [4]
    const float* __restrict__ W_dense,     // [28,64] row-major (k major)
    const float* __restrict__ b_dense,     // [64]
    const float* __restrict__ prelu_a,     // [1]
    const float* __restrict__ ln_w,        // [64]
    const float* __restrict__ ln_b,        // [64]
    float*       __restrict__ out,         // [N,64]
    int n_tiles)                           // N / 16
{
  const int lane = threadIdx.x & 31;
  const int m    = lane & 15;   // row within A-tile / column within D-tile
  const int half = lane >> 4;   // lane group: selects K slice per ISA layout

  // ---------------- per-wave constants (amortized over the tile loop) -------
  const float alpha = prelu_a[0];

  // B matrices: W_dense as 4 N-tiles of 32x16 f16 (K 28..31 zero-padded).
  // Lane layout: col c = 16*t + (lane%16); element e holds K = 8*half + e (e<8), 16 + 8*half + (e-8) (e>=8).
  v16h Bt[4];
#pragma unroll
  for (int t = 0; t < 4; ++t) {
    const int c = t * 16 + m;
#pragma unroll
    for (int e = 0; e < 16; ++e) {
      const int k  = (e < 8) ? (half * 8 + e) : (16 + half * 8 + (e - 8));
      const int kc = (k < 28) ? k : 27;            // clamp (no divergence)
      float w = W_dense[kc * 64 + c];
      w = (k < 28) ? w : 0.0f;                     // zero K padding
      Bt[t][e] = (_Float16)w;
    }
  }

  // Bias folded into the WMMA accumulator: C element v = row (v+8*half),
  // col (16*t + lane%16) -> bias depends only on (lane, t), so a splat per
  // tile t is exactly the b_dense broadcast. Built once, read-only SRC2.
  v8f Cb[4];
  float lw[4], lb[4];
#pragma unroll
  for (int t = 0; t < 4; ++t) {
    const int c = t * 16 + m;
    const float b = b_dense[c];
#pragma unroll
    for (int v = 0; v < 8; ++v) Cb[t][v] = b;
    lw[t] = ln_w[c];
    lb[t] = ln_b[c];
  }

  float won[8], bon[8];
#pragma unroll
  for (int j = 0; j < 8; ++j) { won[j] = W_on[j]; bon[j] = b_on[j]; }
  float wdu[4], bdu[4];
#pragma unroll
  for (int j = 0; j < 4; ++j) { wdu[j] = W_dur[j]; bdu[j] = b_dur[j]; }

  const int wave_id = (blockIdx.x * (blockDim.x >> 5)) + (threadIdx.x >> 5);
  const int n_waves = gridDim.x * (blockDim.x >> 5);

  // ---------------- grid-stride loop over 16-row tiles ----------------------
  for (int tile = wave_id; tile < n_tiles; tile += n_waves) {
    const int row = tile * 16 + m;

    // Streaming per-row inputs (NT: single use, don't pollute caches)
    const int   note = __builtin_nontemporal_load(notes + row);
    const float on   = __builtin_nontemporal_load(onsets + row);
    const float du   = __builtin_nontemporal_load(durs + row);

    // Speculative prefetch of the next tile's streams -> global_prefetch_b8
    {
      const int nrow = row + n_waves * 16;
      __builtin_prefetch(notes + nrow, 0, 1);
      __builtin_prefetch(onsets + nrow, 0, 1);
      __builtin_prefetch(durs + nrow, 0, 1);
    }

    // -------- build A (16x32 f16): x = [note_emb(16) | on(8) | dur(4) | 0] --
    // elements 0..7  <-> K = 8*half .. 8*half+7     (note embedding slice)
    // elements 8..15 <-> K = 16+8*half .. 23+8*half (onset / dur+pad)
    v16h A;
    {
      const float4 nt0 = *(const float4*)(note_table + note * 16 + half * 8);
      const float4 nt1 = *(const float4*)(note_table + note * 16 + half * 8 + 4);
      A[0] = (_Float16)nt0.x; A[1] = (_Float16)nt0.y;
      A[2] = (_Float16)nt0.z; A[3] = (_Float16)nt0.w;
      A[4] = (_Float16)nt1.x; A[5] = (_Float16)nt1.y;
      A[6] = (_Float16)nt1.z; A[7] = (_Float16)nt1.w;
#pragma unroll
      for (int j = 0; j < 8; ++j) {
        const float on_v = fmaf(on, won[j], bon[j]);
        const float du_v = (j < 4) ? fmaf(du, wdu[j], bdu[j]) : 0.0f;
        A[8 + j] = (_Float16)(half ? du_v : on_v);   // v_cndmask, no branch
      }
    }

    // -------- 4 WMMAs: D[t] = A(16x32) x Bt(32x16) + bias, f32 accum -------
    v8f d[4];
#pragma unroll
    for (int t = 0; t < 4; ++t) {
      d[t] = __builtin_amdgcn_wmma_f32_16x16x32_f16(
          /*neg_a=*/false, A, /*neg_b=*/false, Bt[t],
          /*c_mod=*/(short)0, Cb[t], /*reuse_a=*/false, /*reuse_b=*/false);
    }

    // -------- epilogue: PReLU + LayerNorm(64) per row -----------------------
    // D element v = row (v + 8*half), col (16*t + lane%16): the 64 values of a
    // row live in 4 regs x 16 lanes of one half -> butterfly masks 1,2,4,8.
#pragma unroll
    for (int v = 0; v < 8; ++v) {
      float val[4];
      float s = 0.0f, ss = 0.0f;
#pragma unroll
      for (int t = 0; t < 4; ++t) {
        const float x = d[t][v];
        const float p = fmaxf(x, 0.0f) + alpha * fminf(x, 0.0f);  // PReLU
        val[t] = p; s += p; ss += p * p;
      }
#pragma unroll
      for (int mask = 1; mask < 16; mask <<= 1) {   // 16-lane allreduce
        s  += __shfl_xor(s,  mask, 32);
        ss += __shfl_xor(ss, mask, 32);
      }
      const float mu  = s  * (1.0f / 64.0f);
      const float var = ss * (1.0f / 64.0f) - mu * mu;
      const float r   = rsqrtf(var + LN_EPS);
      const int orow  = tile * 16 + half * 8 + v;
#pragma unroll
      for (int t = 0; t < 4; ++t) {                 // NT streaming store
        const float g = r * lw[t];
        const float y = fmaf(val[t] - mu, g, lb[t]);
        __builtin_nontemporal_store(y, out + orow * 64 + t * 16 + m);
      }
    }
  }
}

extern "C" void kernel_launch(void* const* d_in, const int* in_sizes, int n_in,
                              void* d_out, int out_size, void* d_ws, size_t ws_size,
                              hipStream_t stream) {
  const int*   notes      = (const int*)  d_in[0];
  const float* onsets     = (const float*)d_in[1];
  const float* durs       = (const float*)d_in[2];
  const float* note_table = (const float*)d_in[3];
  const float* W_on       = (const float*)d_in[4];
  const float* b_on       = (const float*)d_in[5];
  const float* W_dur      = (const float*)d_in[6];
  const float* b_dur      = (const float*)d_in[7];
  const float* W_dense    = (const float*)d_in[8];
  const float* b_dense    = (const float*)d_in[9];
  const float* prelu_a    = (const float*)d_in[10];
  const float* ln_w       = (const float*)d_in[11];
  const float* ln_b       = (const float*)d_in[12];
  float* out = (float*)d_out;

  const int n_rows  = in_sizes[0];        // B*S = 524288
  const int n_tiles = n_rows / 16;        // 32768

  // 256 threads = 8 wave32 per block; ~4 tiles per wave to amortize the
  // per-wave W_dense->f16 B-register / bias-accumulator build.
  const int threads = 256;
  int blocks = (n_tiles + 31) / 32;       // 8 waves * 4 tiles each
  if (blocks < 1) blocks = 1;
  if (blocks > 4096) blocks = 4096;

  fused_embed_dense_prelu_ln<<<blocks, threads, 0, stream>>>(
      notes, onsets, durs, note_table, W_on, b_on, W_dur, b_dur,
      W_dense, b_dense, prelu_a, ln_w, ln_b, out, n_tiles);
}
